// GatedGCNModel_28956669510068
// MI455X (gfx1250) — matrix operand order repaired and checked by reading, not compiled, not run on hardware
//
#include <hip/hip_runtime.h>
#include <hip/hip_bf16.h>
#include <math.h>

#define NN 50000
#define EE 600000
#define HH 128

typedef __attribute__((ext_vector_type(16))) __bf16 v16bf;
typedef __attribute__((ext_vector_type(8)))  float  v8f;

__device__ __forceinline__ unsigned short f2bf(float f) {
    unsigned int u = __float_as_uint(f);
    return (unsigned short)((u + 0x7FFFu + ((u >> 16) & 1u)) >> 16);
}

__device__ __forceinline__ float sigmoidf_(float v) { return 1.0f / (1.0f + expf(-v)); }

// ---------------- utility kernels ----------------
__global__ void k_zero(float* __restrict__ p, int n) {
    int i = blockIdx.x * blockDim.x + threadIdx.x;
    if (i < n) p[i] = 0.0f;
}

__global__ void k_cvt_bf16(const float* __restrict__ in, unsigned short* __restrict__ out, int n) {
    int i = blockIdx.x * blockDim.x + threadIdx.x;
    if (i < n) out[i] = f2bf(in[i]);
}

// out[j*128+k] = bf16(in[k*128+j]), for 128x128 weight (row-major B -> col-major Bcm)
__global__ void k_cvt_bf16_T128(const float* __restrict__ in, unsigned short* __restrict__ out) {
    int i = blockIdx.x * blockDim.x + threadIdx.x; // 16384 threads
    if (i >= 128 * 128) return;
    int j = i >> 7, k = i & 127;
    out[j * 128 + k] = f2bf(in[k * 128 + j]);
}

// ---------------- bf16 WMMA GEMM: C[N,Kout] = A[N,128] x B[128,Kout] (+bias, optional relu) ----
// A: bf16 row-major [Nrows,128]. Bcm: bf16 [Kout,128] with Bcm[j][k] = B[k][j].
#define LDA 136
#define LDB 136

__global__ __launch_bounds__(256) void k_gemm_bf16(
    const unsigned short* __restrict__ A,
    const unsigned short* __restrict__ Bcm,
    const float* __restrict__ bias,
    float* __restrict__ C,
    int Nrows, int Kout, int relu)
{
    __shared__ unsigned short Alds[128 * LDA];
    __shared__ unsigned short Blds[64 * LDB];

    const int t = threadIdx.x;
    const int rBase = blockIdx.x * 128;
    const int colBase = blockIdx.y * 64;

    // Stage A tile 128x128 bf16 (coalesced b128 loads)
    {
        int row = t >> 1, part = (t & 1) * 64;
        int rowG = rBase + row;
        if (rowG >= Nrows) rowG = Nrows - 1;
        const uint4* src = (const uint4*)(A + (size_t)rowG * 128 + part);
        uint4* dstp = (uint4*)(Alds + row * LDA + part);
#pragma unroll
        for (int j = 0; j < 4; ++j) dstp[j] = src[j];
    }
    // Stage B tile: 64 output columns, each a 128-long K-vector (already col-major in memory)
    {
        int col = t >> 2, part = (t & 3) * 32;
        const uint4* src = (const uint4*)(Bcm + (size_t)(colBase + col) * 128 + part);
        uint4* dstp = (uint4*)(Blds + col * LDB + part);
#pragma unroll
        for (int j = 0; j < 2; ++j) dstp[j] = src[j];
    }
    __syncthreads();

    const int lane = t & 31;
    const int wave = t >> 5;
    const int hl = lane >> 4;    // which half of the wave
    const int ln = lane & 15;

    union Frag { v16bf v; unsigned int u[8]; };

    v8f acc[4];
#pragma unroll
    for (int n0 = 0; n0 < 4; ++n0) {
        float b = bias ? bias[colBase + n0 * 16 + ln] : 0.0f;
#pragma unroll
        for (int e = 0; e < 8; ++e) acc[n0][e] = b;
    }

    const unsigned short* Arow = Alds + (wave * 16 + ln) * LDA;

#pragma unroll
    for (int ks = 0; ks < 4; ++ks) {
        const int K0 = ks * 32;
        // A fragment, 16x32 bf16 per ISA layout: VGPR v<4 -> K=half*8+v*2, v>=4 -> K=16+half*8+(v-4)*2
        Frag a;
#pragma unroll
        for (int vv = 0; vv < 8; ++vv) {
            int kk = (vv < 4) ? (hl * 8 + vv * 2) : (16 + hl * 8 + (vv - 4) * 2);
            a.u[vv] = *(const unsigned int*)(Arow + K0 + kk);
        }
#pragma unroll
        for (int n0 = 0; n0 < 4; ++n0) {
            // B fragment, 32x16 bf16: same K layout as A (v<4 -> K=half*8+v*2, v>=4 -> K=16+half*8+(v-4)*2)
            Frag b;
            const unsigned short* Bcol = Blds + (n0 * 16 + ln) * LDB + K0;
#pragma unroll
            for (int vv = 0; vv < 8; ++vv) {
                int kk = (vv < 4) ? (hl * 8 + vv * 2) : (16 + hl * 8 + (vv - 4) * 2);
                b.u[vv] = *(const unsigned int*)(Bcol + kk);
            }
            acc[n0] = __builtin_amdgcn_wmma_f32_16x16x32_bf16(
                false, a.v, false, b.v, (short)0, acc[n0], false, false);
        }
    }

    // C layout: VGPR e -> row = e + 8*half, col = lane%16 within 16x16 tile
#pragma unroll
    for (int n0 = 0; n0 < 4; ++n0) {
        int col = colBase + n0 * 16 + ln;
#pragma unroll
        for (int e = 0; e < 8; ++e) {
            int row = rBase + wave * 16 + e + 8 * hl;
            if (row < Nrows) {
                float v = acc[n0][e];
                if (relu) v = fmaxf(v, 0.0f);
                C[(size_t)row * Kout + col] = v;
            }
        }
    }
}

// ---------------- edge scatter: agg[dst] += ew * m[src] (one wave per edge) ----------------
__global__ __launch_bounds__(256) void k_scatter(
    const float* __restrict__ m, const int* __restrict__ src, const int* __restrict__ dst,
    const float* __restrict__ ew, float* __restrict__ agg, int E)
{
    int e = blockIdx.x * (blockDim.x >> 5) + (threadIdx.x >> 5);
    if (e >= E) return;
    int lane = threadIdx.x & 31;
    int s = src[e], d = dst[e];
    float w = ew[e];
    const float4* ms = (const float4*)(m + (size_t)s * HH);
    float4 v = ms[lane];
    float* ad = agg + (size_t)d * HH;
    __builtin_prefetch(ad + lane * 4, 1, 0);   // global_prefetch_b8 on destination line
    atomicAdd(ad + lane * 4 + 0, w * v.x);
    atomicAdd(ad + lane * 4 + 1, w * v.y);
    atomicAdd(ad + lane * 4 + 2, w * v.z);
    atomicAdd(ad + lane * 4 + 3, w * v.w);
}

// ---------------- fused GRU cell + ReLU ----------------
__global__ void k_gru_relu(const float* __restrict__ gi, const float* __restrict__ gh,
                           const float* __restrict__ x, float* __restrict__ out, int n)
{
    int idx = blockIdx.x * blockDim.x + threadIdx.x;
    if (idx >= n * HH) return;
    int row = idx >> 7, c = idx & 127;
    const float* gin = gi + (size_t)row * 384;
    const float* ghn = gh + (size_t)row * 384;
    float r = sigmoidf_(gin[c] + ghn[c]);
    float z = sigmoidf_(gin[c + 128] + ghn[c + 128]);
    float nn = tanhf(gin[c + 256] + r * ghn[c + 256]);
    float h = (1.0f - z) * nn + z * x[idx];
    out[idx] = fmaxf(h, 0.0f);
}

// ---------------- BatchNorm: per-channel sum/sumsq then normalize ----------------
__global__ void k_bn_stats(const float* __restrict__ h, float* __restrict__ stats, int n)
{
    int c = threadIdx.x;   // 128 threads = 128 channels, coalesced row reads
    float s = 0.0f, ss = 0.0f;
    for (int r = blockIdx.x; r < n; r += gridDim.x) {
        float v = h[(size_t)r * HH + c];
        s += v; ss += v * v;
    }
    atomicAdd(&stats[c], s);
    atomicAdd(&stats[128 + c], ss);
}

__global__ void k_bn_norm(const float* __restrict__ hin, const float* __restrict__ stats,
                          const float* __restrict__ g, const float* __restrict__ b,
                          float* __restrict__ hout, unsigned short* __restrict__ hbf, int n)
{
    int idx = blockIdx.x * blockDim.x + threadIdx.x;
    if (idx >= n * HH) return;
    int c = idx & 127;
    float invn = 1.0f / (float)n;
    float mean = stats[c] * invn;
    float var = stats[128 + c] * invn - mean * mean;
    float sc = rsqrtf(var + 1e-5f) * g[c];
    float sh = b[c] - mean * sc;
    float v = hin[idx] * sc + sh;
    hout[idx] = v;
    if (hbf) hbf[idx] = f2bf(v);
}

// ---------------- final [N,128] x [128,1] + bias, one wave per row ----------------
__global__ __launch_bounds__(256) void k_final_dot(const float* __restrict__ h,
                                                   const float* __restrict__ lw2,
                                                   const float* __restrict__ lb2,
                                                   float* __restrict__ out, int n)
{
    int row = blockIdx.x * (blockDim.x >> 5) + (threadIdx.x >> 5);
    if (row >= n) return;
    int lane = threadIdx.x & 31;
    const float4* hr = (const float4*)(h + (size_t)row * HH);
    const float4* wr = (const float4*)lw2;
    float4 a = hr[lane], b = wr[lane];
    float s = a.x * b.x + a.y * b.y + a.z * b.z + a.w * b.w;
#pragma unroll
    for (int o = 16; o > 0; o >>= 1) s += __shfl_xor(s, o, 32);
    if (lane == 0) out[row] = s + lb2[0];
}

// ---------------- launcher ----------------
extern "C" void kernel_launch(void* const* d_in, const int* in_sizes, int n_in,
                              void* d_out, int out_size, void* d_ws, size_t ws_size,
                              hipStream_t stream)
{
    const float* x    = (const float*)d_in[0];
    const float* ew   = (const float*)d_in[1];
    const float* W1   = (const float*)d_in[2];
    const float* Wih1 = (const float*)d_in[3];
    const float* Whh1 = (const float*)d_in[4];
    const float* bih1 = (const float*)d_in[5];
    const float* bhh1 = (const float*)d_in[6];
    const float* W2   = (const float*)d_in[7];
    const float* Wih2 = (const float*)d_in[8];
    const float* Whh2 = (const float*)d_in[9];
    const float* bih2 = (const float*)d_in[10];
    const float* bhh2 = (const float*)d_in[11];
    const float* lw1  = (const float*)d_in[12];
    const float* lb1  = (const float*)d_in[13];
    const float* lw2  = (const float*)d_in[14];
    const float* lb2  = (const float*)d_in[15];
    const float* g1   = (const float*)d_in[16];
    const float* be1  = (const float*)d_in[17];
    const float* g2   = (const float*)d_in[18];
    const float* be2  = (const float*)d_in[19];
    const float* g3   = (const float*)d_in[20];
    const float* be3  = (const float*)d_in[21];
    const int*   eidx = (const int*)d_in[22];
    const int*   src  = eidx;
    const int*   dst  = eidx + EE;
    float* out = (float*)d_out;

    const size_t NH  = (size_t)NN * HH;     // 6.4M
    const size_t N3H = (size_t)NN * 3 * HH; // 19.2M

    char* ws = (char*)d_ws;
    size_t off = 0;
    auto salloc = [&](size_t bytes) -> char* {
        char* p = ws + off;
        off += (bytes + 255) & ~(size_t)255;
        return p;
    };
    unsigned short* hbf    = (unsigned short*)salloc(NH * 2);   // current input in bf16 (x, then h1, h2)
    unsigned short* aggbf  = (unsigned short*)salloc(NH * 2);
    float* m    = (float*)salloc(NH * 4);
    float* agg  = (float*)salloc(NH * 4);
    float* gi   = (float*)salloc(N3H * 4);
    float* gh   = (float*)salloc(N3H * 4);
    float* h1   = (float*)salloc(NH * 4);
    unsigned short* W1cm  = (unsigned short*)salloc(128 * 128 * 2);
    unsigned short* W2cm  = (unsigned short*)salloc(128 * 128 * 2);
    unsigned short* Wih1b = (unsigned short*)salloc(384 * 128 * 2);
    unsigned short* Whh1b = (unsigned short*)salloc(384 * 128 * 2);
    unsigned short* Wih2b = (unsigned short*)salloc(384 * 128 * 2);
    unsigned short* Whh2b = (unsigned short*)salloc(384 * 128 * 2);
    unsigned short* lw1b  = (unsigned short*)salloc(128 * 128 * 2);
    float* stats = (float*)salloc(256 * 4);
    float* h2 = agg;  // agg is dead after it is cast to bf16; reuse for h2
    float* h3 = m;    // m is dead after layer-2 scatter; reuse for h3
    (void)ws_size; (void)n_in; (void)in_sizes; (void)out_size;

    const int T = 256;
    const int gNH  = (int)((NH  + T - 1) / T);   // 25000
    const int gN3H = (int)((N3H + T - 1) / T);
    const dim3 gemmG128((NN + 127) / 128, 2);    // Kout=128
    const dim3 gemmG384((NN + 127) / 128, 6);    // Kout=384
    const int gEdges = (EE + 7) / 8;             // 8 edges (waves) per block

    // ---- one-time precision conversion of weights + x ----
    k_cvt_bf16<<<gNH, T, 0, stream>>>(x, hbf, (int)NH);
    k_cvt_bf16_T128<<<64, T, 0, stream>>>(W1, W1cm);
    k_cvt_bf16_T128<<<64, T, 0, stream>>>(W2, W2cm);
    k_cvt_bf16<<<192, T, 0, stream>>>(Wih1, Wih1b, 384 * 128);
    k_cvt_bf16<<<192, T, 0, stream>>>(Whh1, Whh1b, 384 * 128);
    k_cvt_bf16<<<192, T, 0, stream>>>(Wih2, Wih2b, 384 * 128);
    k_cvt_bf16<<<192, T, 0, stream>>>(Whh2, Whh2b, 384 * 128);
    k_cvt_bf16<<<64, T, 0, stream>>>(lw1, lw1b, 128 * 128);

    // ================= layer 1 =================
    k_gemm_bf16<<<gemmG128, T, 0, stream>>>(hbf, W1cm, nullptr, m, NN, 128, 0);
    k_zero<<<gNH, T, 0, stream>>>(agg, (int)NH);
    k_scatter<<<gEdges, T, 0, stream>>>(m, src, dst, ew, agg, EE);
    k_cvt_bf16<<<gNH, T, 0, stream>>>(agg, aggbf, (int)NH);
    k_gemm_bf16<<<gemmG384, T, 0, stream>>>(aggbf, Wih1b, bih1, gi, NN, 384, 0);
    k_gemm_bf16<<<gemmG384, T, 0, stream>>>(hbf, Whh1b, bhh1, gh, NN, 384, 0);
    k_gru_relu<<<gNH, T, 0, stream>>>(gi, gh, x, h1, NN);
    k_zero<<<1, 256, 0, stream>>>(stats, 256);
    k_bn_stats<<<512, 128, 0, stream>>>(h1, stats, NN);
    k_bn_norm<<<gNH, T, 0, stream>>>(h1, stats, g1, be1, h1, hbf, NN);  // hbf now = bf16(h1)

    // ================= layer 2 =================
    k_gemm_bf16<<<gemmG128, T, 0, stream>>>(hbf, W2cm, nullptr, m, NN, 128, 0);
    k_zero<<<gNH, T, 0, stream>>>(agg, (int)NH);
    k_scatter<<<gEdges, T, 0, stream>>>(m, src, dst, ew, agg, EE);
    k_cvt_bf16<<<gNH, T, 0, stream>>>(agg, aggbf, (int)NH);
    k_gemm_bf16<<<gemmG384, T, 0, stream>>>(aggbf, Wih2b, bih2, gi, NN, 384, 0);
    k_gemm_bf16<<<gemmG384, T, 0, stream>>>(hbf, Whh2b, bhh2, gh, NN, 384, 0);
    k_gru_relu<<<gNH, T, 0, stream>>>(gi, gh, h1, h2, NN);
    k_zero<<<1, 256, 0, stream>>>(stats, 256);
    k_bn_stats<<<512, 128, 0, stream>>>(h2, stats, NN);
    k_bn_norm<<<gNH, T, 0, stream>>>(h2, stats, g2, be2, h2, hbf, NN);  // hbf now = bf16(h2)

    // ================= head: relu(h2 @ lw1.T + lb1) -> BN -> dot with lw2 =================
    k_gemm_bf16<<<gemmG128, T, 0, stream>>>(hbf, lw1b, lb1, h3, NN, 128, 1);
    k_zero<<<1, 256, 0, stream>>>(stats, 256);
    k_bn_stats<<<512, 128, 0, stream>>>(h3, stats, NN);
    k_bn_norm<<<gNH, T, 0, stream>>>(h3, stats, g3, be3, h3, nullptr, NN);
    k_final_dot<<<(NN + 7) / 8, T, 0, stream>>>(h3, lw2, lb2, out, NN);
}
